// BiLSTM_CRF_74998718923452
// MI455X (gfx1250) — matrix operand order repaired and checked by reading, not compiled, not run on hardware
//
#include <hip/hip_runtime.h>
#include <hip/hip_bf16.h>
#include <cmath>

typedef __bf16 bf16_t;
typedef __attribute__((ext_vector_type(16))) __bf16 v16bf;
typedef __attribute__((ext_vector_type(8)))  __bf16 v8bf;
typedef __attribute__((ext_vector_type(8)))  float  v8f;
typedef __attribute__((ext_vector_type(4)))  unsigned int u32x4;
typedef __attribute__((ext_vector_type(8)))  int i32x8;
typedef __attribute__((ext_vector_type(4)))  int i32x4;

#define B_      64
#define T_      512
#define E_      256
#define H_      512
#define L_      9
#define BT_     (B_ * T_)
#define G4H     (4 * H_)
#define NSLICE  16      // workgroups per direction in recurrent kernel
#define SLICE_H 32      // h columns owned per workgroup
#define SLICE_G 128     // gate rows owned per workgroup (4 * SLICE_H)

#if defined(__has_builtin)
#if __has_builtin(__builtin_amdgcn_tensor_load_to_lds) && \
    __has_builtin(__builtin_amdgcn_s_wait_tensorcnt)
#define HAVE_TDM 1
#endif
#endif

// ---------------------------------------------------------------------------
// WMMA fragment loaders (contiguous k = 16*(lane/16) + s in both A and B, not the ISA table order)
// ---------------------------------------------------------------------------
__device__ __forceinline__ v16bf load_a_frag(const bf16_t* tile, int ld) {
  int lane = threadIdx.x & 31;
  int row  = lane & 15;
  int koff = (lane >> 4) * 16;
  const bf16_t* p = tile + row * ld + koff;
  v8bf lo = *(const v8bf*)(p);
  v8bf hi = *(const v8bf*)(p + 8);
  return __builtin_shufflevector(lo, hi, 0,1,2,3,4,5,6,7,8,9,10,11,12,13,14,15);
}

__device__ __forceinline__ v16bf load_b_frag(const bf16_t* tile, int ld) {
  int lane = threadIdx.x & 31;
  int col  = lane & 15;
  int koff = (lane >> 4) * 16;
  return *(const v16bf*)(tile + col * ld + koff);
}

__device__ __forceinline__ v8f wmma_bf16(v16bf a, v16bf b, v8f c) {
  return __builtin_amdgcn_wmma_f32_16x16x32_bf16(false, a, false, b, (short)0, c,
                                                 false, false);
}

// ---------------------------------------------------------------------------
// Tensor Data Mover: build D# (CDNA5 ISA) and issue a
// (up to 3-D) bf16 tile load global->LDS. One wave issues; TENSORcnt tracks.
//   element = 2 bytes; tile (x,y,z) = tile0 x tile1 x tile2
//   global elem addr = base + x + y*stride0 + z*stride1
// ---------------------------------------------------------------------------
__device__ __forceinline__ void tdm_load_tile_bf16(
    unsigned lds_off, const void* gptr,
    unsigned tile0, unsigned tile1, unsigned tile2,
    unsigned td0, unsigned td1, unsigned td2,
    unsigned long long s0, unsigned long long s1) {
#if defined(HAVE_TDM)
  unsigned long long ga = (unsigned long long)gptr;
  u32x4 g0;
  g0[0] = 1u;                                            // count=1, user desc
  g0[1] = lds_off;                                       // LDS byte address
  g0[2] = (unsigned)(ga & 0xFFFFFFFFu);                  // global_addr[31:0]
  g0[3] = (unsigned)((ga >> 32) & 0x01FFFFFFu) | (2u << 30);  // addr[56:32]|type=2
  i32x8 g1;
  g1[0] = (int)(1u << 16);                               // data_size=1 (2 bytes)
  g1[1] = (int)((td0 & 0xFFFFu) << 16);                  // tensor_dim0[15:0]
  g1[2] = (int)(((td0 >> 16) & 0xFFFFu) | ((td1 & 0xFFFFu) << 16));
  g1[3] = (int)(((td1 >> 16) & 0xFFFFu) | ((tile0 & 0xFFFFu) << 16));
  g1[4] = (int)((tile1 & 0xFFFFu) | ((tile2 & 0xFFFFu) << 16));
  g1[5] = (int)(s0 & 0xFFFFFFFFull);                     // dim0 stride [31:0]
  g1[6] = (int)(((s0 >> 32) & 0xFFFFull) | ((s1 & 0xFFFFull) << 16));
  g1[7] = (int)((s1 >> 16) & 0xFFFFFFFFull);             // dim1 stride [47:16]
  i32x4 g2 = {(int)td2, 0, 0, 0};                        // tensor_dim2
  i32x4 g3 = {0, 0, 0, 0};
#if defined(__clang_major__) && __clang_major__ >= 23
  i32x8 g4 = {0, 0, 0, 0, 0, 0, 0, 0};
  __builtin_amdgcn_tensor_load_to_lds(g0, g1, g2, g3, g4, 0);
#else
  __builtin_amdgcn_tensor_load_to_lds(g0, g1, g2, g3, 0);
#endif
#else
  (void)lds_off; (void)gptr; (void)tile0; (void)tile1; (void)tile2;
  (void)td0; (void)td1; (void)td2; (void)s0; (void)s1;
#endif
}

__device__ __forceinline__ void tdm_wait() {
#if defined(HAVE_TDM)
  __builtin_amdgcn_s_wait_tensorcnt(0);
#endif
}

// ---------------------------------------------------------------------------
// Prep kernels
// ---------------------------------------------------------------------------
__global__ void f2bf_kernel(const float* __restrict__ in, bf16_t* __restrict__ out,
                            int n) {
  int i = blockIdx.x * 256 + threadIdx.x;
  if (i < n) out[i] = (bf16_t)in[i];
}

__global__ void bias_kernel(const float* __restrict__ bih,
                            const float* __restrict__ bhh,
                            float* __restrict__ out) {
  int i = blockIdx.x * 256 + threadIdx.x;
  if (i < G4H) out[i] = bih[i] + bhh[i];
}

// one block per (b,t) row; 256 threads = E columns
__global__ void embed_kernel(const int* __restrict__ x,
                             const float* __restrict__ table,
                             bf16_t* __restrict__ out) {
  int row = blockIdx.x;
  int tok = x[row];
  out[(size_t)row * E_ + threadIdx.x] =
      (bf16_t)table[(size_t)tok * E_ + threadIdx.x];
}

__global__ void init_kernel(bf16_t* __restrict__ hbuf, unsigned* __restrict__ sync) {
  int i = blockIdx.x * 256 + threadIdx.x;
  if (i < 2 * 2 * B_ * H_) hbuf[i] = (bf16_t)0.0f;
  if (i < 2) sync[i] = 0u;
}

// ---------------------------------------------------------------------------
// Input projection: xg[dir] = emb @ Wih[dir]^T + (bih+bhh)   [BT, 4H] f32
// grid = (BT/128, 4H/64, 2), block = 256 (8 waves).
// ---------------------------------------------------------------------------
__global__ __launch_bounds__(256) void gemm_xg_kernel(
    const bf16_t* __restrict__ emb,   // [BT, E]
    const bf16_t* __restrict__ wih,   // [2][4H, E]
    const float*  __restrict__ bias,  // [2][4H]
    float*        __restrict__ xg)    // [2][BT, 4H]
{
  int wave = threadIdx.x >> 5;
  int lane = threadIdx.x & 31;
  int m0   = blockIdx.x * 128 + wave * 16;
  int nb   = blockIdx.y * 64;
  int dir  = blockIdx.z;

  const bf16_t* W  = wih + (size_t)dir * G4H * E_;
  const float*  bs = bias + (size_t)dir * G4H;
  float*        out = xg + (size_t)dir * BT_ * G4H;

  v8f acc[4] = {v8f{}, v8f{}, v8f{}, v8f{}};
  for (int k0 = 0; k0 < E_; k0 += 32) {
    v16bf a = load_a_frag(emb + (size_t)m0 * E_ + k0, E_);
#pragma unroll
    for (int s = 0; s < 4; ++s) {
      v16bf b = load_b_frag(W + (size_t)(nb + s * 16) * E_ + k0, E_);
      acc[s] = wmma_bf16(a, b, acc[s]);
    }
  }

  int nn = lane & 15;
  int mo = (lane >> 4) * 8;
#pragma unroll
  for (int s = 0; s < 4; ++s) {
    int n = nb + s * 16 + nn;
    float bv = bs[n];
#pragma unroll
    for (int r = 0; r < 8; ++r) {
      out[(size_t)(m0 + mo + r) * G4H + n] = acc[s][r] + bv;
    }
  }
}

// ---------------------------------------------------------------------------
// Persistent cooperative BiLSTM recurrence.
// 32 WGs: dir = bid/16, slice s = bid%16 owns h cols [32s,32s+32) and gate rows
// {q*512 + 32s + jj}. Whh slice (128x512 bf16 = 128KB) loaded to LDS once via
// TDM (3-D tile: 4 chunks x 32 rows x 512 cols). Per step: TDM-stage h_prev
// (64KB) to LDS, WMMA GEMM from LDS, gates->LDS, pointwise cell update,
// prefetch next step's xg, per-direction grid sync.
// LDS map: [0,128K) Whh | [128K,192K) h | [192K,224K) gates | [224K,232K) c
// ---------------------------------------------------------------------------
#define LDS_W_OFF 0u
#define LDS_H_OFF 131072u
#define LSTM_LDS_BYTES (131072 + 65536 + 32768 + 8192)

__global__ __launch_bounds__(256, 1) void lstm_kernel(
    const float*  __restrict__ xg,    // [2][BT, 4H]
    const bf16_t* __restrict__ whh,   // [2][4H, H]
    const float*  __restrict__ bias,  // [2][4H]
    bf16_t*       __restrict__ hbuf,  // [2 dir][2 phase][B, H]
    float*        __restrict__ hout,  // [BT, 2H]
    unsigned*     __restrict__ sync_ctr)  // [2]
{
  extern __shared__ char smem[];
  bf16_t* lds_w = (bf16_t*)smem;                       // [128][512] 128KB
  bf16_t* lds_h = (bf16_t*)(smem + 131072);            // [64][512]   64KB
  float*  lds_g = (float*)(smem + 131072 + 65536);     // [64][128]   32KB
  float*  lds_c = (float*)(smem + 131072 + 65536 + 32768); // [64][32] 8KB

  int bid  = blockIdx.x;
  int dir  = bid / NSLICE;
  int s    = bid % NSLICE;
  int tid  = threadIdx.x;
  int wave = tid >> 5;
  int lane = tid & 31;

  const float*  xg_d   = xg + (size_t)dir * BT_ * G4H;
  const bf16_t* whh_d  = whh + (size_t)dir * G4H * H_;
  const float*  bias_d = bias + (size_t)dir * G4H;
  bf16_t*       hb     = hbuf + (size_t)dir * 2 * B_ * H_;

  // --- Preload Whh slice into LDS ---
#if defined(HAVE_TDM)
  if (wave == 0) {
    // 3-D tile: x = 512 cols, y = 32 rows (stride 512), z = 4 gate chunks
    // (stride 512*512); base = row s*32 of gate chunk 0.
    tdm_load_tile_bf16(LDS_W_OFF, whh_d + (size_t)s * SLICE_H * H_,
                       /*tile*/ H_, SLICE_H, 4,
                       /*tdim*/ H_, SLICE_H, 4,
                       /*strides*/ (unsigned long long)H_,
                       (unsigned long long)H_ * (unsigned long long)H_);
    tdm_wait();
  }
#else
  for (int idx = tid; idx < SLICE_G * (H_ / 8); idx += 256) {
    int r  = idx / (H_ / 8);
    int kc = (idx % (H_ / 8)) * 8;
    int grow = (r >> 5) * H_ + s * SLICE_H + (r & 31);
    *(v8bf*)(&lds_w[r * H_ + kc]) = *(const v8bf*)(&whh_d[(size_t)grow * H_ + kc]);
  }
#endif
  for (int idx = tid; idx < B_ * SLICE_H; idx += 256) lds_c[idx] = 0.0f;
  __syncthreads();

  int ph = 0;
  for (int step = 0; step < T_; ++step) {
    int t = dir ? (T_ - 1 - step) : step;
    const bf16_t* hprev = hb + (size_t)ph * B_ * H_;

    // --- stage h_prev (64x512 bf16, contiguous 64KB) into LDS ---
#if defined(HAVE_TDM)
    if (wave == 0) {
      tdm_load_tile_bf16(LDS_H_OFF, hprev,
                         /*tile*/ (unsigned)(B_ * H_), 0, 0,
                         /*tdim*/ (unsigned)(B_ * H_), 1, 1,
                         (unsigned long long)(B_ * H_),
                         (unsigned long long)(B_ * H_));
      tdm_wait();
    }
#else
    for (int idx = tid; idx < (B_ * H_) / 8; idx += 256) {
      *(v8bf*)(&lds_h[idx * 8]) = *(const v8bf*)(&hprev[idx * 8]);
    }
#endif
    __syncthreads();

    // --- GEMM: [64,512] x [512,128]^T from LDS. Wave w: N strip 16w, 4 M-tiles.
    int n0 = wave * 16;
    v8f acc[4] = {v8f{}, v8f{}, v8f{}, v8f{}};
    for (int k0 = 0; k0 < H_; k0 += 32) {
      v16bf bfrag = load_b_frag(&lds_w[(size_t)n0 * H_ + k0], H_);
#pragma unroll
      for (int mt = 0; mt < 4; ++mt) {
        v16bf afrag = load_a_frag(&lds_h[(size_t)(mt * 16) * H_ + k0], H_);
        acc[mt] = wmma_bf16(afrag, bfrag, acc[mt]);
      }
    }

    // --- add xg + bias, scatter pre-activation gates to LDS; prefetch t+1 ---
    int nn   = lane & 15;
    int mo   = (lane >> 4) * 8;
    int gcol = n0 + nn;                                      // 0..127 local
    int grow = (gcol >> 5) * H_ + s * SLICE_H + (gcol & 31); // global gate row
    float bv = bias_d[grow];
    int t2 = dir ? (T_ - 2 - step) : (step + 1);
    bool do_pf = (step + 1 < T_);
#pragma unroll
    for (int mt = 0; mt < 4; ++mt) {
#pragma unroll
      for (int r = 0; r < 8; ++r) {
        int b = mt * 16 + mo + r;
        float g = acc[mt][r] + bv + xg_d[((size_t)b * T_ + t) * G4H + grow];
        lds_g[b * SLICE_G + gcol] = g;
        if (do_pf)
          __builtin_prefetch(&xg_d[((size_t)b * T_ + t2) * G4H + grow], 0, 1);
      }
    }
    __syncthreads();

    // --- pointwise LSTM cell update on owned 64 x 32 slice ---
    bf16_t* hnext = hb + (size_t)(ph ^ 1) * B_ * H_;
    for (int idx = tid; idx < B_ * SLICE_H; idx += 256) {
      int b  = idx >> 5;
      int jj = idx & 31;
      float gi = lds_g[b * SLICE_G + jj];
      float gf = lds_g[b * SLICE_G + 32 + jj];
      float gg = lds_g[b * SLICE_G + 64 + jj];
      float go = lds_g[b * SLICE_G + 96 + jj];
      float si = 1.0f / (1.0f + __expf(-gi));
      float sf = 1.0f / (1.0f + __expf(-gf));
      float so = 1.0f / (1.0f + __expf(-go));
      float c  = sf * lds_c[idx] + si * tanhf(gg);
      float h  = so * tanhf(c);
      lds_c[idx] = c;
      int j = s * SLICE_H + jj;
      hnext[b * H_ + j] = (bf16_t)h;
      hout[((size_t)b * T_ + t) * (2 * H_) + dir * H_ + j] = h;
    }

    // --- per-direction grid sync ---
    __threadfence();
    __syncthreads();
    if (tid == 0) {
      __hip_atomic_fetch_add(&sync_ctr[dir], 1u, __ATOMIC_RELEASE,
                             __HIP_MEMORY_SCOPE_AGENT);
      unsigned target = (unsigned)(step + 1) * NSLICE;
      while (__hip_atomic_load(&sync_ctr[dir], __ATOMIC_ACQUIRE,
                               __HIP_MEMORY_SCOPE_AGENT) < target) {
        __builtin_amdgcn_s_sleep(2);
      }
    }
    __syncthreads();
    ph ^= 1;
  }
}

// ---------------------------------------------------------------------------
// FC: emissions[b,t,l] = hout[b,t,:] . fcW[l,:] + fcb[l]   (wave per row)
// ---------------------------------------------------------------------------
__global__ __launch_bounds__(256) void fc_kernel(const float* __restrict__ hout,
                                                 const float* __restrict__ fcW,
                                                 const float* __restrict__ fcb,
                                                 float* __restrict__ emis) {
  int gw   = (blockIdx.x * 256 + threadIdx.x) >> 5;
  int lane = threadIdx.x & 31;
  if (gw >= BT_) return;
  const float* hr = hout + (size_t)gw * (2 * H_);
  float hv[32];
#pragma unroll
  for (int i = 0; i < 32; ++i) hv[i] = hr[lane + 32 * i];
#pragma unroll 1
  for (int l = 0; l < L_; ++l) {
    const float* wr = fcW + (size_t)l * (2 * H_);
    float acc = 0.0f;
#pragma unroll
    for (int i = 0; i < 32; ++i) acc += hv[i] * wr[lane + 32 * i];
    for (int off = 16; off > 0; off >>= 1) acc += __shfl_xor(acc, off, 32);
    if (lane == 0) emis[(size_t)gw * L_ + l] = acc + fcb[l];
  }
}

// ---------------------------------------------------------------------------
// CRF NLL: single workgroup, 576 threads = (b,j) pairs; sequential over T.
// ---------------------------------------------------------------------------
__global__ __launch_bounds__(576) void crf_kernel(const float* __restrict__ emis,
                                                  const int* __restrict__ tags,
                                                  const float* __restrict__ start_t,
                                                  const float* __restrict__ end_t,
                                                  const float* __restrict__ trans,
                                                  float* __restrict__ out) {
  __shared__ float sc[B_][L_];
  __shared__ float tr[L_][L_];
  __shared__ float num[B_];
  __shared__ float red[B_];
  int tid = threadIdx.x;
  int b = tid / L_;
  int j = tid % L_;
  if (tid < L_ * L_) tr[tid / L_][tid % L_] = trans[tid];
  sc[b][j] = start_t[j] + emis[(size_t)b * T_ * L_ + j];
  __syncthreads();

  if (j == 0) {  // numerator (gold path score) per batch element
    int prev = tags[b * T_];
    float n = start_t[prev] + end_t[tags[b * T_ + T_ - 1]] +
              emis[(size_t)b * T_ * L_ + prev];
    for (int t = 1; t < T_; ++t) {
      int cur = tags[b * T_ + t];
      n += emis[((size_t)b * T_ + t) * L_ + cur] + tr[prev][cur];
      prev = cur;
    }
    num[b] = n;
  }

  for (int t = 1; t < T_; ++t) {  // forward recursion (log-sum-exp)
    float m = -3.0e38f;
#pragma unroll
    for (int i = 0; i < L_; ++i) m = fmaxf(m, sc[b][i] + tr[i][j]);
    float ssum = 0.0f;
#pragma unroll
    for (int i = 0; i < L_; ++i) ssum += __expf(sc[b][i] + tr[i][j] - m);
    float v = m + __logf(ssum) + emis[((size_t)b * T_ + t) * L_ + j];
    __syncthreads();
    sc[b][j] = v;
    __syncthreads();
  }

  if (j == 0) {
    float m = -3.0e38f;
#pragma unroll
    for (int i = 0; i < L_; ++i) m = fmaxf(m, sc[b][i] + end_t[i]);
    float ssum = 0.0f;
#pragma unroll
    for (int i = 0; i < L_; ++i) ssum += __expf(sc[b][i] + end_t[i] - m);
    red[b] = num[b] - (m + __logf(ssum));
  }
  __syncthreads();
  if (tid == 0) {
    float acc = 0.0f;
    for (int bb = 0; bb < B_; ++bb) acc += red[bb];
    out[0] = -acc;
  }
}

// ---------------------------------------------------------------------------
// Host launcher
// ---------------------------------------------------------------------------
extern "C" void kernel_launch(void* const* d_in, const int* in_sizes, int n_in,
                              void* d_out, int out_size, void* d_ws, size_t ws_size,
                              hipStream_t stream) {
  const int*   x         = (const int*)d_in[0];
  const int*   tags      = (const int*)d_in[1];
  const float* emb_table = (const float*)d_in[2];
  const float* Wih_f     = (const float*)d_in[3];
  const float* Whh_f     = (const float*)d_in[4];
  const float* bih_f     = (const float*)d_in[5];
  const float* bhh_f     = (const float*)d_in[6];
  const float* Wih_b     = (const float*)d_in[7];
  const float* Whh_b     = (const float*)d_in[8];
  const float* bih_b     = (const float*)d_in[9];
  const float* bhh_b     = (const float*)d_in[10];
  const float* fcW       = (const float*)d_in[11];
  const float* fcb       = (const float*)d_in[12];
  const float* start_t   = (const float*)d_in[13];
  const float* end_t     = (const float*)d_in[14];
  const float* trans     = (const float*)d_in[15];
  float* outp = (float*)d_out;

  char* ws = (char*)d_ws;
  size_t off = 0;
  auto alloc = [&](size_t bytes) {
    size_t r = off;
    off = (off + bytes + 255) & ~(size_t)255;
    return r;
  };
  size_t o_emb  = alloc((size_t)BT_ * E_ * 2);       // emb bf16
  size_t o_wih  = alloc((size_t)2 * G4H * E_ * 2);   // Wih bf16 [2]
  size_t o_whh  = alloc((size_t)2 * G4H * H_ * 2);   // Whh bf16 [2]
  size_t o_bias = alloc((size_t)2 * G4H * 4);        // combined bias f32 [2]
  size_t o_xg   = alloc((size_t)2 * BT_ * G4H * 4);  // gate preactivations f32 [2]
  size_t o_hbuf = alloc((size_t)2 * 2 * B_ * H_ * 2);// h exchange bf16 [2dir][2ph]
  size_t o_hout = alloc((size_t)BT_ * 2 * H_ * 4);   // hidden states f32
  size_t o_emis = alloc((size_t)BT_ * L_ * 4);       // emissions f32
  size_t o_sync = alloc(256);                        // sync counters
  (void)ws_size; (void)in_sizes; (void)n_in; (void)out_size;

  bf16_t*   emb_bf  = (bf16_t*)(ws + o_emb);
  bf16_t*   wih_bf  = (bf16_t*)(ws + o_wih);
  bf16_t*   whh_bf  = (bf16_t*)(ws + o_whh);
  float*    bias    = (float*)(ws + o_bias);
  float*    xg      = (float*)(ws + o_xg);
  bf16_t*   hbuf    = (bf16_t*)(ws + o_hbuf);
  float*    hout    = (float*)(ws + o_hout);
  float*    emis    = (float*)(ws + o_emis);
  unsigned* syncp   = (unsigned*)(ws + o_sync);

  // 1) weight conversion + bias combine
  int nIh = G4H * E_, nHh = G4H * H_;
  f2bf_kernel<<<(nIh + 255) / 256, 256, 0, stream>>>(Wih_f, wih_bf, nIh);
  f2bf_kernel<<<(nIh + 255) / 256, 256, 0, stream>>>(Wih_b, wih_bf + nIh, nIh);
  f2bf_kernel<<<(nHh + 255) / 256, 256, 0, stream>>>(Whh_f, whh_bf, nHh);
  f2bf_kernel<<<(nHh + 255) / 256, 256, 0, stream>>>(Whh_b, whh_bf + nHh, nHh);
  bias_kernel<<<(G4H + 255) / 256, 256, 0, stream>>>(bih_f, bhh_f, bias);
  bias_kernel<<<(G4H + 255) / 256, 256, 0, stream>>>(bih_b, bhh_b, bias + G4H);

  // 2) embedding gather (f32 -> bf16)
  embed_kernel<<<BT_, E_, 0, stream>>>(x, emb_table, emb_bf);

  // 3) input projection GEMM (WMMA bf16)
  gemm_xg_kernel<<<dim3(BT_ / 128, G4H / 64, 2), 256, 0, stream>>>(emb_bf, wih_bf,
                                                                   bias, xg);

  // 4) init h exchange buffers + sync counters, then persistent recurrence
  init_kernel<<<(2 * 2 * B_ * H_ + 255) / 256, 256, 0, stream>>>(hbuf, syncp);
  lstm_kernel<<<2 * NSLICE, 256, LSTM_LDS_BYTES, stream>>>(xg, whh_bf, bias, hbuf,
                                                           hout, syncp);

  // 5) FC emissions, 6) CRF NLL
  fc_kernel<<<BT_ / 8, 256, 0, stream>>>(hout, fcW, fcb, emis);
  crf_kernel<<<1, B_ * L_, 0, stream>>>(emis, tags, start_t, end_t, trans, outp);
}
